// TransformerXLRPE_53712861004262
// MI455X (gfx1250) — compile-verified, matrix operand order checked by reading, not run on hardware
//
#include <hip/hip_runtime.h>
#include <hip/hip_bf16.h>
#include <cstdint>

// ---------------------------------------------------------------------------
// Transformer-XL relative-position attention scores for MI455X (gfx1250).
// All GEMMs on v_wmma_f32_16x16x32_bf16 (fp32 accumulate).
// Double-buffered LDS fed by GLOBAL_LOAD_ASYNC_TO_LDS_B128 (ASYNCcnt),
// one barrier per 32-K step, buffer parity compile-time (K-loop unrolled x2),
// immediate-offset ds_loads, 8 WMMAs / wave / K-step.
// ---------------------------------------------------------------------------

typedef __bf16 bf16;
typedef __attribute__((ext_vector_type(8)))  __bf16 v8bf;
typedef __attribute__((ext_vector_type(16))) __bf16 v16bf;
typedef __attribute__((ext_vector_type(8)))  float  v8f;

#define BB   4
#define LL   2048
#define DD   1024
#define MREL 512
#define TAB  1025          // 2*MREL+1
#define NP   1152          // TAB padded to multiple of 128
#define BL   (BB*LL)       // 8192

// ---- workspace layout (bytes) ---------------------------------------------
#define OFF_XB   ((size_t)0)                         // bf16 x        (8192x1024)
#define OFF_WQ   (OFF_XB  + (size_t)BL*DD*2)         // bf16 Wq       (1024x1024)
#define OFF_WK   (OFF_WQ  + (size_t)DD*DD*2)
#define OFF_WR   (OFF_WK  + (size_t)DD*DD*2)
#define OFF_TB   (OFF_WR  + (size_t)DD*DD*2)         // bf16 table    (1152x1024, padded)
#define OFF_QU   (OFF_TB  + (size_t)NP*DD*2)         // bf16 Q+u      (8192x1024)
#define OFF_QV   (OFF_QU  + (size_t)BL*DD*2)         // bf16 Q+v
#define OFF_KB   (OFF_QV  + (size_t)BL*DD*2)         // bf16 K
#define OFF_RB   (OFF_KB  + (size_t)BL*DD*2)         // bf16 R        (1152x1024)
#define OFF_AS   (OFF_RB  + (size_t)NP*DD*2)         // f32  A scores (8192x1152)

__device__ __forceinline__ v16bf cat8(v8bf lo, v8bf hi) {
    return __builtin_shufflevector(lo, hi,
        0,1,2,3,4,5,6,7,8,9,10,11,12,13,14,15);
}

__device__ __forceinline__ v16bf ldfrag(const bf16* p) {
    return cat8(*(const v8bf*)p, *(const v8bf*)(p + 16));
}

#define WMMA_BF16(A, B, C) __builtin_amdgcn_wmma_f32_16x16x32_bf16( \
    false, (A), false, (B), (short)0, (C), false, false)

// Async 16B copy global -> LDS (GLOBAL_LOAD_ASYNC_TO_LDS_B128, ASYNCcnt).
// ldsoff: workgroup LDS byte address (low 32 bits of generic LDS pointer,
// ISA 10.2: LDS aperture maps addr[31:0] to the LDS offset).
__device__ __forceinline__ void async_cp16(uint32_t ldsoff, const bf16* src) {
    asm volatile("global_load_async_to_lds_b128 %0, %1, off"
                 :: "v"(ldsoff), "v"((uint64_t)(uintptr_t)src)
                 : "memory");
}

__device__ __forceinline__ void wait_async0() {
    asm volatile("s_wait_asynccnt 0x0" ::: "memory");
}

// Issue all 12 fragment loads, then the 8-WMMA chain (single dscnt wait).
__device__ __forceinline__ void compute_step(const bf16* rdA0, const bf16* rdA1,
                                             const bf16* rdB, v8f acc[2][4])
{
    v16bf a0 = ldfrag(rdA0);
    v16bf a1 = ldfrag(rdA1);
    v16bf b0 = ldfrag(rdB);
    v16bf b1 = ldfrag(rdB + 16 * 32);
    v16bf b2 = ldfrag(rdB + 32 * 32);
    v16bf b3 = ldfrag(rdB + 48 * 32);
    acc[0][0] = WMMA_BF16(a0, b0, acc[0][0]);
    acc[1][0] = WMMA_BF16(a1, b0, acc[1][0]);
    acc[0][1] = WMMA_BF16(a0, b1, acc[0][1]);
    acc[1][1] = WMMA_BF16(a1, b1, acc[1][1]);
    acc[0][2] = WMMA_BF16(a0, b2, acc[0][2]);
    acc[1][2] = WMMA_BF16(a1, b2, acc[1][2]);
    acc[0][3] = WMMA_BF16(a0, b3, acc[0][3]);
    acc[1][3] = WMMA_BF16(a1, b3, acc[1][3]);
}

// ---------------------------------------------------------------------------
// 128x64 output tile: C[m,n] = sum_k A[m,k]*B[n,k], A/B bf16 row-major.
// 128 threads (4 waves); wave w owns rows w*32..w*32+31 (2 A-frags x 4 B-frags).
// Double-buffered LDS: sA[2][128*32] (8KB each), sB[2][64*32] (4KB each).
// ---------------------------------------------------------------------------
__device__ __forceinline__ void gemm128x64(const bf16* __restrict__ Abase, int ldA,
                                           const bf16* __restrict__ Bbase, int ldB,
                                           int Klen, v8f acc[2][4],
                                           bf16* sA, bf16* sB)
{
    const int tid  = threadIdx.x;
    const int wave = tid >> 5;
    const int lane = tid & 31;
    const int r    = lane & 15;         // frag row (A) / col (B)
    const int h    = lane >> 4;         // K-half select

    const uint32_t sAo = (uint32_t)(uintptr_t)sA;
    const uint32_t sBo = (uint32_t)(uintptr_t)sB;

    // ---- per-thread staging addresses, computed once ----------------------
    const bf16* gA[4];  uint32_t lA0[4], lA1[4];
    #pragma unroll
    for (int c = 0; c < 4; ++c) {                  // A: 512 16B chunks
        int chunk = c * 128 + tid, row = chunk >> 2, col8 = (chunk & 3) * 8;
        gA[c]  = Abase + (size_t)row * ldA + col8;
        lA0[c] = sAo + (uint32_t)(row * 64 + col8 * 2);
        lA1[c] = lA0[c] + 8192;
    }
    const bf16* gB[2];  uint32_t lB0[2], lB1[2];
    #pragma unroll
    for (int c = 0; c < 2; ++c) {                  // B: 256 16B chunks
        int chunk = c * 128 + tid, row = chunk >> 2, col8 = (chunk & 3) * 8;
        gB[c]  = Bbase + (size_t)row * ldB + col8;
        lB0[c] = sBo + (uint32_t)(row * 64 + col8 * 2);
        lB1[c] = lB0[c] + 4096;
    }

    // ---- fixed per-thread LDS fragment read bases -------------------------
    const bf16* rdA0 = sA + (wave * 32 + r) * 32 + h * 8;
    const bf16* rdA1 = rdA0 + 16 * 32;
    const bf16* rdB  = sB + r * 32 + h * 8;

    #pragma unroll
    for (int a = 0; a < 2; ++a)
        #pragma unroll
        for (int j = 0; j < 4; ++j) acc[a][j] = (v8f){};

    // ---- prologue: stage k=0 into buffer 0 --------------------------------
    #pragma unroll
    for (int c = 0; c < 4; ++c) async_cp16(lA0[c], gA[c]);
    #pragma unroll
    for (int c = 0; c < 2; ++c) async_cp16(lB0[c], gB[c]);

    const int steps = Klen >> 6;        // two 32-K halves per step
    for (int kk = 0; kk < steps; ++kk) {
        // ---- half 0: compute buffer 0, prefetch into buffer 1 -------------
        wait_async0();                  // my writes to buf0 landed
        __syncthreads();                // everyone's; all done reading buf1
        #pragma unroll
        for (int c = 0; c < 4; ++c) { gA[c] += 32; async_cp16(lA1[c], gA[c]); }
        #pragma unroll
        for (int c = 0; c < 2; ++c) { gB[c] += 32; async_cp16(lB1[c], gB[c]); }
        compute_step(rdA0, rdA1, rdB, acc);

        // ---- half 1: compute buffer 1, prefetch into buffer 0 -------------
        wait_async0();
        __syncthreads();
        if (kk + 1 < steps) {
            #pragma unroll
            for (int c = 0; c < 4; ++c) { gA[c] += 32; async_cp16(lA0[c], gA[c]); }
            #pragma unroll
            for (int c = 0; c < 2; ++c) { gB[c] += 32; async_cp16(lB0[c], gB[c]); }
        }
        compute_step(rdA0 + 4096, rdA1 + 4096, rdB + 2048, acc);
    }
}

// ---------------------------------------------------------------------------
// Stage 1: casts / padding (8 elements per thread, b128 traffic)
// ---------------------------------------------------------------------------
__global__ __launch_bounds__(256) void cast_bf16x8_kernel(const float* __restrict__ in,
                                                          bf16* __restrict__ out, int n8)
{
    int i = blockIdx.x * blockDim.x + threadIdx.x;
    if (i >= n8) return;
    const float4* in4 = (const float4*)in;
    float4 f0 = in4[(size_t)i * 2];
    float4 f1 = in4[(size_t)i * 2 + 1];
    v8bf o = { (bf16)f0.x, (bf16)f0.y, (bf16)f0.z, (bf16)f0.w,
               (bf16)f1.x, (bf16)f1.y, (bf16)f1.z, (bf16)f1.w };
    *(v8bf*)(out + (size_t)i * 8) = o;
}

__global__ __launch_bounds__(256) void pad_table_kernel(const float* __restrict__ tab,
                                                        bf16* __restrict__ out)
{
    int i = blockIdx.x * blockDim.x + threadIdx.x;   // over NP*DD/8
    if (i >= NP * DD / 8) return;
    int row = (i * 8) >> 10;                         // DD = 1024
    v8bf o;
    if (row < TAB) {
        const float4* in4 = (const float4*)tab;
        float4 f0 = in4[(size_t)i * 2];
        float4 f1 = in4[(size_t)i * 2 + 1];
        o = (v8bf){ (bf16)f0.x, (bf16)f0.y, (bf16)f0.z, (bf16)f0.w,
                    (bf16)f1.x, (bf16)f1.y, (bf16)f1.z, (bf16)f1.w };
    } else {
        o = (v8bf){};
    }
    *(v8bf*)(out + (size_t)i * 8) = o;
}

// ---------------------------------------------------------------------------
// Stage 2: Q/K projection.  z==0: writes bf16(Q+u), bf16(Q+v); z==1: bf16(K)
// ---------------------------------------------------------------------------
__global__ __launch_bounds__(128) void gemm_qk_kernel(const bf16* __restrict__ xb,
                                                      const bf16* __restrict__ wqb,
                                                      const bf16* __restrict__ wkb,
                                                      const float* __restrict__ u,
                                                      const float* __restrict__ v,
                                                      bf16* __restrict__ qub,
                                                      bf16* __restrict__ qvb,
                                                      bf16* __restrict__ kb)
{
    __shared__ __align__(16) bf16 sA[2 * 128 * 32];
    __shared__ __align__(16) bf16 sB[2 * 64 * 32];
    const int m0 = blockIdx.x * 128;                 // over BL
    const int n0 = blockIdx.y * 64;                  // over DD
    const bf16* W = blockIdx.z ? wkb : wqb;

    v8f acc[2][4];
    gemm128x64(xb + (size_t)m0 * DD, DD, W + (size_t)n0 * DD, DD, DD, acc, sA, sB);

    const int lane = threadIdx.x & 31, wave = threadIdx.x >> 5;
    const int r = lane & 15, h = lane >> 4;
    #pragma unroll
    for (int j = 0; j < 4; ++j) {
        int col = n0 + j * 16 + r;
        float uu = u[col], vv = v[col];
        #pragma unroll
        for (int a = 0; a < 2; ++a) {
            #pragma unroll
            for (int i = 0; i < 8; ++i) {
                size_t row = (size_t)(m0 + wave * 32 + a * 16 + h * 8 + i);
                size_t off = row * DD + col;
                float q = acc[a][j][i];
                if (blockIdx.z == 0) {
                    qub[off] = (bf16)(q + uu);
                    qvb[off] = (bf16)(q + vv);
                } else {
                    kb[off] = (bf16)q;
                }
            }
        }
    }
}

// ---------------------------------------------------------------------------
// Stage 3: R = table @ Wr^T  -> bf16 (NP x DD)
// ---------------------------------------------------------------------------
__global__ __launch_bounds__(128) void gemm_r_kernel(const bf16* __restrict__ tabb,
                                                     const bf16* __restrict__ wrb,
                                                     bf16* __restrict__ rb)
{
    __shared__ __align__(16) bf16 sA[2 * 128 * 32];
    __shared__ __align__(16) bf16 sB[2 * 64 * 32];
    const int m0 = blockIdx.x * 128;                 // over NP
    const int n0 = blockIdx.y * 64;                  // over DD

    v8f acc[2][4];
    gemm128x64(tabb + (size_t)m0 * DD, DD, wrb + (size_t)n0 * DD, DD, DD, acc, sA, sB);

    const int lane = threadIdx.x & 31, wave = threadIdx.x >> 5;
    const int r = lane & 15, h = lane >> 4;
    #pragma unroll
    for (int j = 0; j < 4; ++j) {
        int col = n0 + j * 16 + r;
        #pragma unroll
        for (int a = 0; a < 2; ++a)
            #pragma unroll
            for (int i = 0; i < 8; ++i) {
                size_t row = (size_t)(m0 + wave * 32 + a * 16 + h * 8 + i);
                rb[row * DD + col] = (bf16)acc[a][j][i];
            }
    }
}

// ---------------------------------------------------------------------------
// Stage 4: A = (Q+v) @ R^T  -> f32 (BL x NP)
// ---------------------------------------------------------------------------
__global__ __launch_bounds__(128) void gemm_a_kernel(const bf16* __restrict__ qvb,
                                                     const bf16* __restrict__ rb,
                                                     float* __restrict__ asc)
{
    __shared__ __align__(16) bf16 sA[2 * 128 * 32];
    __shared__ __align__(16) bf16 sB[2 * 64 * 32];
    const int m0 = blockIdx.x * 128;                 // over BL
    const int n0 = blockIdx.y * 64;                  // over NP

    v8f acc[2][4];
    gemm128x64(qvb + (size_t)m0 * DD, DD, rb + (size_t)n0 * DD, DD, DD, acc, sA, sB);

    const int lane = threadIdx.x & 31, wave = threadIdx.x >> 5;
    const int r = lane & 15, h = lane >> 4;
    #pragma unroll
    for (int j = 0; j < 4; ++j) {
        int col = n0 + j * 16 + r;
        #pragma unroll
        for (int a = 0; a < 2; ++a)
            #pragma unroll
            for (int i = 0; i < 8; ++i) {
                size_t row = (size_t)(m0 + wave * 32 + a * 16 + h * 8 + i);
                asc[row * NP + col] = acc[a][j][i];
            }
    }
}

// ---------------------------------------------------------------------------
// Stage 5: content = (Q+u) @ K^T per batch, fused gather + scale -> out
// ---------------------------------------------------------------------------
__global__ __launch_bounds__(128) void scores_kernel(const bf16* __restrict__ qub,
                                                     const bf16* __restrict__ kb,
                                                     const float* __restrict__ asc,
                                                     float* __restrict__ out)
{
    __shared__ __align__(16) bf16 sA[2 * 128 * 32];
    __shared__ __align__(16) bf16 sB[2 * 64 * 32];
    const int b  = blockIdx.z;
    const int i0 = blockIdx.y * 128;
    const int j0 = blockIdx.x * 64;
    const bf16* Abase = qub + (size_t)b * LL * DD + (size_t)i0 * DD;
    const bf16* Bbase = kb  + (size_t)b * LL * DD + (size_t)j0 * DD;

    v8f acc[2][4];
    gemm128x64(Abase, DD, Bbase, DD, DD, acc, sA, sB);

    const float scale = 0.03125f;                    // 1/sqrt(1024)
    const int lane = threadIdx.x & 31, wave = threadIdx.x >> 5;
    const int r = lane & 15, h = lane >> 4;
    #pragma unroll
    for (int j = 0; j < 4; ++j) {
        int col = j0 + j * 16 + r;
        #pragma unroll
        for (int a = 0; a < 2; ++a)
            #pragma unroll
            for (int i = 0; i < 8; ++i) {
                int row = i0 + wave * 32 + a * 16 + h * 8 + i;
                int dlt = row - col;
                dlt = dlt < -MREL ? -MREL : (dlt > MREL ? MREL : dlt);
                int pos = dlt + MREL;                // in [0, 1024]
                size_t arow = (size_t)b * LL + row;
                float ps = asc[arow * NP + pos];
                out[arow * LL + col] = (acc[a][j][i] + ps) * scale;
            }
    }
}

// ---------------------------------------------------------------------------
extern "C" void kernel_launch(void* const* d_in, const int* in_sizes, int n_in,
                              void* d_out, int out_size, void* d_ws, size_t ws_size,
                              hipStream_t stream)
{
    const float* x   = (const float*)d_in[0];
    const float* Wq  = (const float*)d_in[1];
    const float* Wk  = (const float*)d_in[2];
    const float* Wr  = (const float*)d_in[3];
    const float* u   = (const float*)d_in[4];
    const float* v   = (const float*)d_in[5];
    const float* tab = (const float*)d_in[6];
    float* out = (float*)d_out;

    char* ws = (char*)d_ws;
    bf16*  xb   = (bf16*)(ws + OFF_XB);
    bf16*  wqb  = (bf16*)(ws + OFF_WQ);
    bf16*  wkb  = (bf16*)(ws + OFF_WK);
    bf16*  wrb  = (bf16*)(ws + OFF_WR);
    bf16*  tabb = (bf16*)(ws + OFF_TB);
    bf16*  qub  = (bf16*)(ws + OFF_QU);
    bf16*  qvb  = (bf16*)(ws + OFF_QV);
    bf16*  kbuf = (bf16*)(ws + OFF_KB);
    bf16*  rb   = (bf16*)(ws + OFF_RB);
    float* asc  = (float*)(ws + OFF_AS);

    // Stage 1: casts / padding (vectorized x8)
    {
        int n8 = BL * DD / 8;
        cast_bf16x8_kernel<<<(n8 + 255) / 256, 256, 0, stream>>>(x, xb, n8);
        n8 = DD * DD / 8;
        cast_bf16x8_kernel<<<(n8 + 255) / 256, 256, 0, stream>>>(Wq, wqb, n8);
        cast_bf16x8_kernel<<<(n8 + 255) / 256, 256, 0, stream>>>(Wk, wkb, n8);
        cast_bf16x8_kernel<<<(n8 + 255) / 256, 256, 0, stream>>>(Wr, wrb, n8);
        n8 = NP * DD / 8;
        pad_table_kernel<<<(n8 + 255) / 256, 256, 0, stream>>>(tab, tabb);
    }

    // Stage 2: Q/K projection (z=0: Q+u / Q+v, z=1: K)
    gemm_qk_kernel<<<dim3(BL / 128, DD / 64, 2), 128, 0, stream>>>(
        xb, wqb, wkb, u, v, qub, qvb, kbuf);

    // Stage 3: R = table @ Wr^T
    gemm_r_kernel<<<dim3(NP / 128, DD / 64), 128, 0, stream>>>(tabb, wrb, rb);

    // Stage 4: A = (Q+v) @ R^T
    gemm_a_kernel<<<dim3(BL / 128, NP / 64), 128, 0, stream>>>(qvb, rb, asc);

    // Stage 5: content + gathered positional scores -> out
    scores_kernel<<<dim3(LL / 64, LL / 128, BB), 128, 0, stream>>>(
        qub, kbuf, asc, out);
}
